// VarianceAdaptor_33870112096381
// MI455X (gfx1250) — matrix operand order repaired and checked by reading, not compiled, not run on hardware
//
#include <hip/hip_runtime.h>
#include <stdint.h>

#define N_      16
#define T_      1024
#define E_      1024
#define H_      256
#define K_      3
#define MAXOUT_ 4096
#define LN_EPS_ 1e-5f

// padded LDS row strides (halfwords): stride%? -> 4-dword bank shift per row
#define A1_STRIDE 1032   // conv1: 1024 data + 8 pad
#define A2_STRIDE 264    // conv2: 256 data + 8 pad

typedef unsigned short u16;
typedef __attribute__((ext_vector_type(4)))  u16    u16x4;
typedef __attribute__((ext_vector_type(8)))  u16    u16x8;
typedef __attribute__((ext_vector_type(16))) __bf16 v16bf;
typedef __attribute__((ext_vector_type(8)))  float  v8f;

struct U16x16 { u16x8 lo, hi; };

__device__ __forceinline__ u16 f32_to_bf16(float f) {
  union { float f; uint32_t u; } v; v.f = f;
  uint32_t u = v.u;
  uint32_t r = u + 0x7FFFu + ((u >> 16) & 1u);  // round-to-nearest-even
  return (u16)(r >> 16);
}

// async 16B copy global -> LDS (gfx1250 ASYNCcnt path)
__device__ __forceinline__ void async_b128(uint32_t lds_off, const void* gp) {
  uint64_t ga = (uint64_t)(uintptr_t)gp;
  asm volatile("global_load_async_to_lds_b128 %0, %1, off"
               :: "v"(lds_off), "v"(ga) : "memory");
}
__device__ __forceinline__ void wait_async0() {
  asm volatile("s_wait_asynccnt 0x0" ::: "memory");
}
// low 32 bits of a flat LDS pointer == LDS byte address (ISA aperture rule)
__device__ __forceinline__ uint32_t lds_addr(const void* p) {
  return (uint32_t)(uintptr_t)p;
}

// ---------------- prep: masked seqs -> bf16 -------------------------------
__global__ __launch_bounds__(256)
void prep_x_k(const float* __restrict__ seqs, const int* __restrict__ pm,
              u16* __restrict__ xbf) {
  size_t g = (size_t)blockIdx.x * 256 + threadIdx.x;   // float4 index
  size_t lin = g * 4;
  if (lin >= (size_t)N_ * T_ * E_) return;
  int nt = (int)(lin / E_);                             // n*T + t (E % 4 == 0)
  float m = (pm[nt] != 0) ? 1.f : 0.f;
  float4 v = *(const float4*)(seqs + lin);
  u16x4 o;
  o[0] = f32_to_bf16(v.x * m); o[1] = f32_to_bf16(v.y * m);
  o[2] = f32_to_bf16(v.z * m); o[3] = f32_to_bf16(v.w * m);
  *(u16x4*)(xbf + lin) = o;
}

// ---------------- prep: weights (H,E,K) -> [k][e][h] bf16 -----------------
__global__ __launch_bounds__(256)
void prep_w_k(const float* __restrict__ w, u16* __restrict__ wt, int EIN) {
  int o = blockIdx.x * 256 + threadIdx.x;
  if (o >= K_ * EIN * H_) return;
  int h = o % H_;
  int e = (o / H_) % EIN;
  int k = o / (H_ * EIN);
  wt[o] = f32_to_bf16(w[((size_t)h * EIN + e) * K_ + k]);
}

// ---------------- prep: zero halo row -------------------------------------
__global__ __launch_bounds__(256)
void zero_k(u16* __restrict__ zrow) {
  u16x4 z = {};
  *(u16x4*)(zrow + threadIdx.x * 4) = z;               // 1024 halfwords
}

// ---------------- conv1 (E=1024 -> H=256) + ReLU + LN1 + mask -------------
__global__ __launch_bounds__(256)
void conv1_wmma(const u16* __restrict__ xbf, const u16* __restrict__ w1t,
                const u16* __restrict__ zrow,
                const float* __restrict__ cb, const float* __restrict__ g,
                const float* __restrict__ be, const int* __restrict__ pm,
                u16* __restrict__ y1) {
  __shared__ u16 atile[18][A1_STRIDE];                  // 18 halo'd time rows
  __shared__ float yt[16][H_ + 4];
  const int n  = blockIdx.x / (T_ / 16);
  const int t0 = (blockIdx.x % (T_ / 16)) * 16;
  const int wave = threadIdx.x >> 5;
  const int lane = threadIdx.x & 31;
  const int laneLo = lane & 15, laneHi = lane >> 4;
  const int h0 = wave * 32, h1 = h0 + 16;

  // ---- stage A tile (rows t0-1 .. t0+16) into LDS with async copies ----
  #pragma unroll
  for (int i = 0; i < 9; ++i) {                         // 18 rows * 128 chunks
    int q = threadIdx.x + 256 * i;
    int row = q >> 7;                                   // /128 (16B chunks/row)
    int cc  = q & 127;
    int t = t0 + row - 1;
    const u16* src = (t >= 0 && t < T_)
                   ? (xbf + ((size_t)n * T_ + t) * E_ + cc * 8)
                   : (zrow + cc * 8);
    async_b128(lds_addr(&atile[row][cc * 8]), src);
  }
  wait_async0();
  __syncthreads();

  v8f acc0 = {}; v8f acc1 = {};
  for (int k = 0; k < K_; ++k) {
    const u16* wk = w1t + (size_t)k * E_ * H_;
    const int ar = laneLo + k;                          // A row in LDS
    for (int c = 0; c < E_ / 32; ++c) {
      const int eb = c * 32;
      u16x8 alo = *(const u16x8*)&atile[ar][eb + 8 * laneHi];
      u16x8 ahi = *(const u16x8*)&atile[ar][eb + 16 + 8 * laneHi];
      U16x16 apk{alo, ahi};
      v16bf a = __builtin_bit_cast(v16bf, apk);
      const u16* brow = wk + (size_t)(eb + 8 * laneHi) * H_ + laneLo;  // B-layout: lane=N col, elems=K (table order)
      U16x16 b0pk, b1pk;
      #pragma unroll
      for (int s = 0; s < 8; ++s) {
        b0pk.lo[s] = brow[(size_t)s * H_ + h0];
        b0pk.hi[s] = brow[(size_t)(16 + s) * H_ + h0];
        b1pk.lo[s] = brow[(size_t)s * H_ + h1];
        b1pk.hi[s] = brow[(size_t)(16 + s) * H_ + h1];
      }
      v16bf bf0 = __builtin_bit_cast(v16bf, b0pk);
      v16bf bf1 = __builtin_bit_cast(v16bf, b1pk);
      acc0 = __builtin_amdgcn_wmma_f32_16x16x32_bf16(false, a, false, bf0, (short)0, acc0, false, false);
      acc1 = __builtin_amdgcn_wmma_f32_16x16x32_bf16(false, a, false, bf1, (short)0, acc1, false, false);
    }
  }

  #pragma unroll
  for (int r = 0; r < 8; ++r) {                         // C-layout: m=r+8*laneHi
    int m = r + 8 * laneHi;
    float v0 = acc0[r] + cb[h0 + laneLo];
    float v1 = acc1[r] + cb[h1 + laneLo];
    yt[m][h0 + laneLo] = v0 > 0.f ? v0 : 0.f;
    yt[m][h1 + laneLo] = v1 > 0.f ? v1 : 0.f;
  }
  __syncthreads();

  for (int rr = 0; rr < 2; ++rr) {                      // LayerNorm over H
    const int row = wave * 2 + rr;
    const int t = t0 + row;
    float s = 0.f, ss = 0.f;
    #pragma unroll
    for (int j = 0; j < H_ / 32; ++j) {
      float v = yt[row][lane + 32 * j];
      s += v; ss += v * v;
    }
    #pragma unroll
    for (int off = 16; off > 0; off >>= 1) {
      s  += __shfl_xor(s,  off, 32);
      ss += __shfl_xor(ss, off, 32);
    }
    float mu = s * (1.f / H_);
    float rs = rsqrtf(ss * (1.f / H_) - mu * mu + LN_EPS_);
    float msk = (pm[n * T_ + t] != 0) ? 1.f : 0.f;
    u16* orow = y1 + ((size_t)n * T_ + t) * H_;
    #pragma unroll
    for (int j = 0; j < H_ / 32; ++j) {
      int h = lane + 32 * j;
      float v = (yt[row][h] - mu) * rs * g[h] + be[h];
      orow[h] = f32_to_bf16(v * msk);
    }
  }
}

// ---------------- conv2 (H->H) + ReLU + LN2 + proj -> durations -----------
__global__ __launch_bounds__(256)
void conv2_wmma(const u16* __restrict__ x, const u16* __restrict__ w2t,
                const u16* __restrict__ zrow,
                const float* __restrict__ cb, const float* __restrict__ g,
                const float* __restrict__ be, const float* __restrict__ pw,
                const float* __restrict__ pbp, const int* __restrict__ pm,
                int* __restrict__ dur) {
  __shared__ u16 atile[18][A2_STRIDE];
  __shared__ float yt[16][H_ + 4];
  const int n  = blockIdx.x / (T_ / 16);
  const int t0 = (blockIdx.x % (T_ / 16)) * 16;
  const int wave = threadIdx.x >> 5;
  const int lane = threadIdx.x & 31;
  const int laneLo = lane & 15, laneHi = lane >> 4;
  const int h0 = wave * 32, h1 = h0 + 16;

  // ---- stage A tile (18 rows x 256 halfwords = 576 16B-chunks) ----
  #pragma unroll
  for (int i = 0; i < 3; ++i) {
    int q = threadIdx.x + 256 * i;
    if (q < 18 * 32) {
      int row = q >> 5;                                 // 32 chunks per row
      int cc  = q & 31;
      int t = t0 + row - 1;
      const u16* src = (t >= 0 && t < T_)
                     ? (x + ((size_t)n * T_ + t) * H_ + cc * 8)
                     : (zrow + cc * 8);
      async_b128(lds_addr(&atile[row][cc * 8]), src);
    }
  }
  wait_async0();
  __syncthreads();

  v8f acc0 = {}; v8f acc1 = {};
  for (int k = 0; k < K_; ++k) {
    const u16* wk = w2t + (size_t)k * H_ * H_;
    const int ar = laneLo + k;
    for (int c = 0; c < H_ / 32; ++c) {
      const int eb = c * 32;
      u16x8 alo = *(const u16x8*)&atile[ar][eb + 8 * laneHi];
      u16x8 ahi = *(const u16x8*)&atile[ar][eb + 16 + 8 * laneHi];
      U16x16 apk{alo, ahi};
      v16bf a = __builtin_bit_cast(v16bf, apk);
      const u16* brow = wk + (size_t)(eb + 8 * laneHi) * H_ + laneLo;
      U16x16 b0pk, b1pk;
      #pragma unroll
      for (int s = 0; s < 8; ++s) {
        b0pk.lo[s] = brow[(size_t)s * H_ + h0];
        b0pk.hi[s] = brow[(size_t)(16 + s) * H_ + h0];
        b1pk.lo[s] = brow[(size_t)s * H_ + h1];
        b1pk.hi[s] = brow[(size_t)(16 + s) * H_ + h1];
      }
      v16bf bf0 = __builtin_bit_cast(v16bf, b0pk);
      v16bf bf1 = __builtin_bit_cast(v16bf, b1pk);
      acc0 = __builtin_amdgcn_wmma_f32_16x16x32_bf16(false, a, false, bf0, (short)0, acc0, false, false);
      acc1 = __builtin_amdgcn_wmma_f32_16x16x32_bf16(false, a, false, bf1, (short)0, acc1, false, false);
    }
  }

  #pragma unroll
  for (int r = 0; r < 8; ++r) {
    int m = r + 8 * laneHi;
    float v0 = acc0[r] + cb[h0 + laneLo];
    float v1 = acc1[r] + cb[h1 + laneLo];
    yt[m][h0 + laneLo] = v0 > 0.f ? v0 : 0.f;
    yt[m][h1 + laneLo] = v1 > 0.f ? v1 : 0.f;
  }
  __syncthreads();

  const float pb = pbp[0];
  for (int rr = 0; rr < 2; ++rr) {
    const int row = wave * 2 + rr;
    const int t = t0 + row;
    float s = 0.f, ss = 0.f;
    #pragma unroll
    for (int j = 0; j < H_ / 32; ++j) {
      float v = yt[row][lane + 32 * j];
      s += v; ss += v * v;
    }
    #pragma unroll
    for (int off = 16; off > 0; off >>= 1) {
      s  += __shfl_xor(s,  off, 32);
      ss += __shfl_xor(ss, off, 32);
    }
    float mu = s * (1.f / H_);
    float rs = rsqrtf(ss * (1.f / H_) - mu * mu + LN_EPS_);
    float dp = 0.f;
    #pragma unroll
    for (int j = 0; j < H_ / 32; ++j) {
      int h = lane + 32 * j;
      float v = (yt[row][h] - mu) * rs * g[h] + be[h];
      dp += v * pw[h];
    }
    #pragma unroll
    for (int off = 16; off > 0; off >>= 1) dp += __shfl_xor(dp, off, 32);
    if (lane == 0) {
      float ld = dp + pb;
      int di = (int)rintf(expf(ld) - 1.0f);             // jnp.round = rint
      if (di < 0) di = 0;
      if (pm[n * T_ + t] == 0) di = 0;
      dur[n * T_ + t] = di;
    }
  }
}

// ---------------- cumsum per batch row ------------------------------------
__global__ __launch_bounds__(1024)
void cumsum_k(const int* __restrict__ dur, int* __restrict__ cum,
              int* __restrict__ lens, float* __restrict__ outF) {
  __shared__ int s[T_];
  const int n = blockIdx.x, tid = threadIdx.x;
  s[tid] = dur[n * T_ + tid];
  __syncthreads();
  for (int off = 1; off < T_; off <<= 1) {
    int add = (tid >= off) ? s[tid - off] : 0;
    __syncthreads();
    s[tid] += add;
    __syncthreads();
  }
  cum[n * T_ + tid] = s[tid];
  if (tid == T_ - 1) {
    lens[n] = s[tid];
    outF[(size_t)N_ * MAXOUT_ * E_ + n] = (float)s[tid];   // seq_lens tail
  }
}

// ---------------- gather / length-regulate --------------------------------
__global__ __launch_bounds__(256)
void gather_k(const float* __restrict__ seqs, const int* __restrict__ cum,
              const int* __restrict__ lens, float* __restrict__ out) {
  const int n = blockIdx.x / MAXOUT_;
  const int p = blockIdx.x % MAXOUT_;
  __shared__ int sidx, svalid;
  if (threadIdx.x == 0) {
    const int* c = cum + n * T_;
    int lo = 0, hi = T_;                          // searchsorted(c, p, 'right')
    while (lo < hi) { int mid = (lo + hi) >> 1; if (c[mid] <= p) lo = mid + 1; else hi = mid; }
    sidx = (lo > T_ - 1) ? (T_ - 1) : lo;
    svalid = (p < lens[n]) ? 1 : 0;
  }
  __syncthreads();
  float4 val = make_float4(0.f, 0.f, 0.f, 0.f);
  if (svalid)
    val = *(const float4*)(seqs + ((size_t)n * T_ + sidx) * E_ + threadIdx.x * 4);
  *(float4*)(out + ((size_t)n * MAXOUT_ + p) * E_ + threadIdx.x * 4) = val;
}

// ---------------- launch ---------------------------------------------------
extern "C" void kernel_launch(void* const* d_in, const int* in_sizes, int n_in,
                              void* d_out, int out_size, void* d_ws, size_t ws_size,
                              hipStream_t stream) {
  const float* seqs = (const float*)d_in[0];
  const float* w1   = (const float*)d_in[1];
  const float* b1   = (const float*)d_in[2];
  const float* g1   = (const float*)d_in[3];
  const float* be1  = (const float*)d_in[4];
  const float* w2   = (const float*)d_in[5];
  const float* b2   = (const float*)d_in[6];
  const float* g2   = (const float*)d_in[7];
  const float* be2  = (const float*)d_in[8];
  const float* pw   = (const float*)d_in[9];
  const float* pb   = (const float*)d_in[10];
  const int*   pm   = (const int*)d_in[11];
  float* out = (float*)d_out;

  char* ws = (char*)d_ws;
  u16* xbf  = (u16*)(ws);                        // 33,554,432 B
  u16* w1t  = (u16*)(ws + 33554432);             //  1,572,864 B
  u16* w2t  = (u16*)(ws + 35127296);             //    393,216 B
  u16* y1   = (u16*)(ws + 35520512);             //  8,388,608 B
  int* dur  = (int*)(ws + 43909120);             //     65,536 B
  int* cum  = (int*)(ws + 43974656);             //     65,536 B
  int* lens = (int*)(ws + 44040192);             //         64 B
  u16* zrow = (u16*)(ws + 44040256);             //      2,048 B

  prep_x_k<<<(N_ * T_ * E_ / 4 + 255) / 256, 256, 0, stream>>>(seqs, pm, xbf);
  prep_w_k<<<(K_ * E_ * H_ + 255) / 256, 256, 0, stream>>>(w1, w1t, E_);
  prep_w_k<<<(K_ * H_ * H_ + 255) / 256, 256, 0, stream>>>(w2, w2t, H_);
  zero_k<<<1, 256, 0, stream>>>(zrow);

  conv1_wmma<<<N_ * (T_ / 16), 256, 0, stream>>>(xbf, w1t, zrow, b1, g1, be1, pm, y1);
  conv2_wmma<<<N_ * (T_ / 16), 256, 0, stream>>>(y1, w2t, zrow, b2, g2, be2, pw, pb, pm, dur);

  cumsum_k<<<N_, T_, 0, stream>>>(dur, cum, lens, out);
  gather_k<<<N_ * MAXOUT_, 256, 0, stream>>>(seqs, cum, lens, out);
}
